// EnBaseLayer_24507083391546
// MI455X (gfx1250) — matrix operand order repaired and checked by reading, not compiled, not run on hardware
//
#include <hip/hip_runtime.h>
#include <hip/hip_bf16.h>

typedef __bf16 bf16;
typedef float v8f   __attribute__((ext_vector_type(8)));
typedef bf16  v16bf __attribute__((ext_vector_type(16)));
typedef bf16  v8bf  __attribute__((ext_vector_type(8)));

#define N_NODES 50000
#define N_EDGES 800000
#define HID 128

// ---- weight fragment layout (bf16, in d_ws) ----
//   n = ntile*16 + (lane&15),  k = kstep*32 + (lane>>4)*8 + (i&7) + (i>>3)*16   (B 32x16 layout)
// We1: K=288 (280 padded) -> 9 ksteps, We2: 4, Wn1 (K=256): 8, Wn2: 4
#define FRAG_WE1_OFF 0
#define FRAG_WE2_OFF 36864      // 9*8*512
#define FRAG_WN1_OFF 53248      // +4*8*512
#define FRAG_WN2_OFF 86016      // +8*8*512
#define FRAG_TOTAL   102400     // +4*8*512

// ws layout (bytes)
#define MI_OFF    262144ul                      // f32 [N_NODES][128]  (25.6 MB)
#define HBF_OFF   (MI_OFF + 25600000ul)         // bf16 [N_NODES][128] (12.8 MB)
#define EABF_OFF  (HBF_OFF + 12800000ul)        // bf16 [N_EDGES][24]  (38.4 MB)

__global__ __launch_bounds__(256) void pack_weights(
    const float* __restrict__ We1, const float* __restrict__ We2,
    const float* __restrict__ Wn1, const float* __restrict__ Wn2,
    bf16* __restrict__ frag) {
  int idx = blockIdx.x * 256 + threadIdx.x;
  if (idx >= FRAG_TOTAL) return;
  const float* W; int Ksrc, base;
  if (idx < FRAG_WE2_OFF)      { W = We1; Ksrc = 280; base = FRAG_WE1_OFF; }
  else if (idx < FRAG_WN1_OFF) { W = We2; Ksrc = 128; base = FRAG_WE2_OFF; }
  else if (idx < FRAG_WN2_OFF) { W = Wn1; Ksrc = 256; base = FRAG_WN1_OFF; }
  else                         { W = Wn2; Ksrc = 128; base = FRAG_WN2_OFF; }
  int local = idx - base;
  int kstep = local >> 12;
  int rem   = local & 4095;
  int ntile = rem >> 9;
  int l     = rem & 511;
  int lane  = l >> 4, i = l & 15;
  int n = ntile * 16 + (lane & 15);
  int k = kstep * 32 + ((lane >> 4) << 3) + (i & 7) + ((i >> 3) << 4);
  float v = (k < Ksrc) ? W[k * HID + n] : 0.0f;
  frag[idx] = (bf16)v;
}

// zero mi + bf16-convert h and edge_attr (one pass, bandwidth-bound)
__global__ __launch_bounds__(256) void prep_data(
    const float* __restrict__ h, const float* __restrict__ ea,
    float* __restrict__ mi, bf16* __restrict__ hbf, bf16* __restrict__ eabf) {
  long i = (long)blockIdx.x * 256 + threadIdx.x;     // grid covers 25,600,000
  if (i < 6400000) { hbf[i] = (bf16)h[i]; mi[i] = 0.0f; }
  else             { long j = i - 6400000; eabf[j] = (bf16)ea[j]; }
}

// x row = [edge_attr(24) | h[dst](128) | h[src](128) | pad(8)]; every 8-run of
// K lies inside one source (24/152/280 are multiples of 8) -> one 16B load
__device__ __forceinline__ v8bf gather8bf(int kr, const bf16* __restrict__ ar,
                                          const bf16* __restrict__ hi,
                                          const bf16* __restrict__ hj) {
  if (kr < 24)  return *(const v8bf*)(ar + kr);
  if (kr < 152) return *(const v8bf*)(hi + (kr - 24));
  if (kr < 280) return *(const v8bf*)(hj + (kr - 152));
  v8bf z;
#pragma unroll
  for (int i = 0; i < 8; ++i) z[i] = (bf16)0.0f;
  return z;
}

__device__ __forceinline__ v16bf gather16bf(int kb, const bf16* __restrict__ ar,
                                            const bf16* __restrict__ hi,
                                            const bf16* __restrict__ hj) {
  union { v8bf h[2]; v16bf w; } u;
  u.h[0] = gather8bf(kb, ar, hi, hj);
  u.h[1] = gather8bf(kb + 16, ar, hi, hj);
  return u.w;
}

__device__ __forceinline__ v8f wmma_bf16(v16bf a, v16bf b, v8f c) {
  return __builtin_amdgcn_wmma_f32_16x16x32_bf16(false, a, false, b, (short)0, c,
                                                 false, false);
}

// bias+relu -> gate -> scatter-atomics for one 16-row subtile held in D layout
__device__ __forceinline__ void edge_epilogue(
    const v8f* __restrict__ d, int e0s, int lane, const int* __restrict__ eidx,
    const float* __restrict__ be2, const float* __restrict__ Wg, float bg0,
    float* __restrict__ mi) {
  const int half = lane >> 4, c = lane & 15;
  float mij[8][8];
#pragma unroll
  for (int t = 0; t < 8; ++t) {
    int n = t * 16 + c; float b2 = be2[n];
#pragma unroll
    for (int r = 0; r < 8; ++r) {
      float v = d[t][r] + b2; mij[t][r] = v > 0.f ? v : 0.f;
    }
  }
  float gate[8];
#pragma unroll
  for (int r = 0; r < 8; ++r) {                    // row m = r + 8*half
    float s = 0.f;
#pragma unroll
    for (int t = 0; t < 8; ++t) s += mij[t][r] * Wg[t * 16 + c];
    for (int o = 1; o < 16; o <<= 1) s += __shfl_xor(s, o, 16);
    gate[r] = 1.0f / (1.0f + __expf(-(s + bg0)));
  }
#pragma unroll
  for (int r = 0; r < 8; ++r) {
    int dd = eidx[e0s + r + 8 * half];
    float g = gate[r];
    float* row = mi + (long)dd * HID;
#pragma unroll
    for (int t = 0; t < 8; ++t) atomicAdd(row + t * 16 + c, mij[t][r] * g);
  }
}

__global__ __launch_bounds__(256) void edge_kernel(
    const bf16* __restrict__ hbf, const int* __restrict__ eidx,
    const bf16* __restrict__ eabf, const float* __restrict__ be1,
    const float* __restrict__ be2, const float* __restrict__ Wg,
    const float* __restrict__ bg, const bf16* __restrict__ frag,
    float* __restrict__ mi) {
  __shared__ __align__(16) bf16 stage[8][32 * HID];  // per-wave 32x128 bf16
  const int waveInBlk = threadIdx.x >> 5;
  const int lane = threadIdx.x & 31;
  const int half = lane >> 4, c = lane & 15;
  const int e0 = (blockIdx.x * 8 + waveInBlk) * 32;  // 32 edges per wave
  const int m = c;
  const bf16 *ar0, *hi0, *hj0, *ar1, *hi1, *hj1;
  {
    int ea = e0 + m, eb = e0 + 16 + m;
    ar0 = eabf + (long)ea * 24; ar1 = eabf + (long)eb * 24;
    hi0 = hbf + (long)eidx[ea] * HID;            hi1 = hbf + (long)eidx[eb] * HID;
    hj0 = hbf + (long)eidx[N_EDGES + ea] * HID;  hj1 = hbf + (long)eidx[N_EDGES + eb] * HID;
  }
  const v16bf* B = (const v16bf*)frag;

  // ---- GEMM1: [32x288] x [288x128]; each B fragment feeds 2 WMMAs ----
  v8f acc0[8] = {}, acc1[8] = {};
#pragma unroll
  for (int ks = 0; ks < 9; ++ks) {
    int kb = ks * 32 + half * 8;
    v16bf a0 = gather16bf(kb, ar0, hi0, hj0);
    v16bf a1 = gather16bf(kb, ar1, hi1, hj1);
    const v16bf* Bk = B + (FRAG_WE1_OFF >> 4) + ks * 256 + lane;
#pragma unroll
    for (int nt = 0; nt < 8; ++nt) {
      v16bf b = Bk[nt * 32];
      acc0[nt] = wmma_bf16(a0, b, acc0[nt]);
      acc1[nt] = wmma_bf16(a1, b, acc1[nt]);
    }
  }

  // bias + relu -> LDS stage (relayout D -> A fragments)
  bf16* st = stage[waveInBlk];
#pragma unroll
  for (int t = 0; t < 8; ++t) {
    int n = t * 16 + c; float b1 = be1[n];
#pragma unroll
    for (int r = 0; r < 8; ++r) {
      float v0 = acc0[t][r] + b1; v0 = v0 > 0.f ? v0 : 0.f;
      float v1 = acc1[t][r] + b1; v1 = v1 > 0.f ? v1 : 0.f;
      st[(r + 8 * half) * HID + n]      = (bf16)v0;
      st[(16 + r + 8 * half) * HID + n] = (bf16)v1;
    }
  }

  // ---- GEMM2: [32x128] x [128x128] ----
  v8f d0[8] = {}, d1[8] = {};
#pragma unroll
  for (int ks = 0; ks < 4; ++ks) {
    int kb = ks * 32 + half * 8;
    union { v8bf h[2]; v16bf w; } ua, ub;
    ua.h[0] = *(const v8bf*)(st + m * HID + kb);
    ua.h[1] = *(const v8bf*)(st + m * HID + kb + 16);
    ub.h[0] = *(const v8bf*)(st + (16 + m) * HID + kb);
    ub.h[1] = *(const v8bf*)(st + (16 + m) * HID + kb + 16);
    const v16bf* Bk = B + (FRAG_WE2_OFF >> 4) + ks * 256 + lane;
#pragma unroll
    for (int nt = 0; nt < 8; ++nt) {
      v16bf b = Bk[nt * 32];
      d0[nt] = wmma_bf16(ua.w, b, d0[nt]);
      d1[nt] = wmma_bf16(ub.w, b, d1[nt]);
    }
  }

  const float bg0 = bg[0];
  edge_epilogue(d0, e0, lane, eidx, be2, Wg, bg0, mi);
  edge_epilogue(d1, e0 + 16, lane, eidx, be2, Wg, bg0, mi);
}

__global__ __launch_bounds__(256) void node_kernel(
    const bf16* __restrict__ hbf, const float* __restrict__ mi,
    const float* __restrict__ bn1, const float* __restrict__ bn2,
    const bf16* __restrict__ frag, float* __restrict__ out) {
  __shared__ __align__(16) bf16 stage[8][16 * HID];
  const int waveInBlk = threadIdx.x >> 5;
  const int lane = threadIdx.x & 31;
  const int wave = blockIdx.x * 8 + waveInBlk;
  if (wave >= N_NODES / 16) return;        // uniform per wave: EXEC stays all-1
  const int half = lane >> 4, c = lane & 15;
  const int t0 = wave * 16;
  const int m = c;
  const float* mirow = mi  + (long)(t0 + m) * HID;   // f32 (atomic-accumulated)
  const bf16*  hrow  = hbf + (long)(t0 + m) * HID;   // bf16
  const v16bf* B = (const v16bf*)frag;

  // ---- GEMM1: y=[mi|h] [16x256] x Wn1 [256x128] ----
  v8f acc[8] = {};
#pragma unroll
  for (int ks = 0; ks < 8; ++ks) {
    union { v8bf h[2]; v16bf w; } u;
#pragma unroll
    for (int p = 0; p < 2; ++p) {
      int kb = ks * 32 + half * 8 + p * 16;
      if (kb < HID) {
        const float4* q = (const float4*)(mirow + kb);
        float4 x = q[0], y = q[1];
        u.h[p][0]=(bf16)x.x; u.h[p][1]=(bf16)x.y; u.h[p][2]=(bf16)x.z; u.h[p][3]=(bf16)x.w;
        u.h[p][4]=(bf16)y.x; u.h[p][5]=(bf16)y.y; u.h[p][6]=(bf16)y.z; u.h[p][7]=(bf16)y.w;
      } else {
        u.h[p] = *(const v8bf*)(hrow + kb - HID);
      }
    }
    const v16bf* Bk = B + (FRAG_WN1_OFF >> 4) + ks * 256 + lane;
#pragma unroll
    for (int nt = 0; nt < 8; ++nt) acc[nt] = wmma_bf16(u.w, Bk[nt * 32], acc[nt]);
  }

  bf16* st = stage[waveInBlk];
#pragma unroll
  for (int t = 0; t < 8; ++t) {
    int n = t * 16 + c; float b1 = bn1[n];
#pragma unroll
    for (int r = 0; r < 8; ++r) {
      float v = acc[t][r] + b1; v = v > 0.f ? v : 0.f;
      st[(r + 8 * half) * HID + n] = (bf16)v;
    }
  }

  // ---- GEMM2: [16x128] x Wn2 [128x128] ----
  v8f acc2[8] = {};
#pragma unroll
  for (int ks = 0; ks < 4; ++ks) {
    int kb = ks * 32 + half * 8;
    union { v8bf h[2]; v16bf w; } u;
    u.h[0] = *(const v8bf*)(st + m * HID + kb);
    u.h[1] = *(const v8bf*)(st + m * HID + kb + 16);
    const v16bf* Bk = B + (FRAG_WN2_OFF >> 4) + ks * 256 + lane;
#pragma unroll
    for (int nt = 0; nt < 8; ++nt) acc2[nt] = wmma_bf16(u.w, Bk[nt * 32], acc2[nt]);
  }

#pragma unroll
  for (int t = 0; t < 8; ++t) {
    int n = t * 16 + c; float b2 = bn2[n];
#pragma unroll
    for (int r = 0; r < 8; ++r) {
      int node_o = t0 + r + 8 * half;
      out[(long)node_o * HID + n] = acc2[t][r] + b2;
    }
  }
}

extern "C" void kernel_launch(void* const* d_in, const int* in_sizes, int n_in,
                              void* d_out, int out_size, void* d_ws, size_t ws_size,
                              hipStream_t stream) {
  const float* h     = (const float*)d_in[0];
  const int*   eidx  = (const int*)d_in[1];
  const float* eattr = (const float*)d_in[2];
  const float* We1   = (const float*)d_in[3];
  const float* be1   = (const float*)d_in[4];
  const float* We2   = (const float*)d_in[5];
  const float* be2   = (const float*)d_in[6];
  const float* Wg    = (const float*)d_in[7];
  const float* bg    = (const float*)d_in[8];
  const float* Wn1   = (const float*)d_in[9];
  const float* bn1   = (const float*)d_in[10];
  const float* Wn2   = (const float*)d_in[11];
  const float* bn2   = (const float*)d_in[12];

  bf16*  frag = (bf16*)d_ws;
  float* mi   = (float*)((char*)d_ws + MI_OFF);
  bf16*  hbf  = (bf16*)((char*)d_ws + HBF_OFF);
  bf16*  eabf = (bf16*)((char*)d_ws + EABF_OFF);
  float* out  = (float*)d_out;

  pack_weights<<<(FRAG_TOTAL + 255) / 256, 256, 0, stream>>>(We1, We2, Wn1, Wn2, frag);
  prep_data<<<25600000 / 256, 256, 0, stream>>>(h, eattr, mi, hbf, eabf);
  edge_kernel<<<N_EDGES / 32 / 8, 256, 0, stream>>>(hbf, eidx, eabf, be1, be2, Wg,
                                                    bg, frag, mi);
  node_kernel<<<(N_NODES / 16 + 7) / 8, 256, 0, stream>>>(hbf, mi, bn1, bn2, frag, out);
}
